// HybridAtlasBlock_42090679501214
// MI455X (gfx1250) — matrix operand order repaired and checked by reading, not compiled, not run on hardware
//
#include <hip/hip_runtime.h>

// ---------------------------------------------------------------- constants
#define B_   2
#define T_   2048
#define D_   1024
#define H_   16
#define DH_  64
#define D2_  32
#define RS_  256
#define R2_  512
#define BT_  (B_*T_)

typedef __attribute__((ext_vector_type(16))) __bf16 v16bf;
typedef __attribute__((ext_vector_type(8)))  float  v8f;
typedef int v4i_t __attribute__((vector_size(16)));   // matches builtin param type

#if __has_builtin(__builtin_amdgcn_global_load_async_to_lds_b128)
#define HAVE_ASYNC_LDS 1
#else
#define HAVE_ASYNC_LDS 0
#endif

__device__ __forceinline__ unsigned short f2bf(float f) {
  unsigned u = __float_as_uint(f);
  u += 0x7FFFu + ((u >> 16) & 1u);          // round-to-nearest-even
  return (unsigned short)(u >> 16);
}

// A-layout fragment (16-bit 16x32): k = 16*hi + e.
// caller passes base = row_ptr + k0 + 16*hi
__device__ __forceinline__ void load_a16(v16bf& a, const unsigned short* base) {
  uint4* ap = (uint4*)&a;
  ap[0] = *(const uint4*)(base);
  ap[1] = *(const uint4*)(base + 8);
}
// B-layout fragment (16-bit 32x16): k = 16*hi + e.
// caller passes base = col_ptr + k0 + 16*hi
__device__ __forceinline__ void load_b16(v16bf& b, const unsigned short* base) {
  uint4* bp = (uint4*)&b;
  bp[0] = *(const uint4*)(base);
  bp[1] = *(const uint4*)(base + 8);
}

// async global->LDS copy of 16 bytes (8 bf16)
__device__ __forceinline__ void cp_g2l_16B(const unsigned short* g, unsigned short* l) {
#if HAVE_ASYNC_LDS
  unsigned short* gnc = const_cast<unsigned short*>(g);
  __attribute__((address_space(1))) v4i_t* gp =
      (__attribute__((address_space(1))) v4i_t*)gnc;
  __attribute__((address_space(3))) v4i_t* lp =
      (__attribute__((address_space(3))) v4i_t*)l;
  __builtin_amdgcn_global_load_async_to_lds_b128(gp, lp, 0, 0);
#else
  *(uint4*)l = *(const uint4*)g;
#endif
}
template <int N>
__device__ __forceinline__ void wait_async_n() {
#if HAVE_ASYNC_LDS
#if __has_builtin(__builtin_amdgcn_s_wait_asynccnt)
  __builtin_amdgcn_s_wait_asynccnt(N);
#else
  if (N == 0) asm volatile("s_wait_asynccnt 0" ::: "memory");
  else        asm volatile("s_wait_asynccnt 4" ::: "memory");
#endif
#endif
}

// ---------------------------------------------------------------- f32 -> bf16
__global__ void __launch_bounds__(256) cvt_bf16_kernel(const float* __restrict__ src,
                                                       unsigned short* __restrict__ dst,
                                                       int n) {
  for (int i = blockIdx.x * 256 + threadIdx.x; i < n; i += gridDim.x * 256)
    dst[i] = f2bf(src[i]);
}

// ---------------------------------------------------------------- RMSNorm -> bf16
__global__ void __launch_bounds__(256) rmsnorm_bf16_kernel(const float* __restrict__ x,
                                                           const float* __restrict__ scale,
                                                           unsigned short* __restrict__ out,
                                                           int cols) {
  __shared__ float red[256];
  const int row = blockIdx.x;
  const float* xr = x + (size_t)row * cols;
  float s = 0.f;
  for (int i = threadIdx.x; i < cols; i += 256) { float v = xr[i]; s += v * v; }
  red[threadIdx.x] = s; __syncthreads();
  for (int off = 128; off > 0; off >>= 1) {
    if (threadIdx.x < off) red[threadIdx.x] += red[threadIdx.x + off];
    __syncthreads();
  }
  const float inv = rsqrtf(red[0] / (float)cols + 1e-6f);
  unsigned short* orow = out + (size_t)row * cols;
  for (int i = threadIdx.x; i < cols; i += 256) orow[i] = f2bf(xr[i] * inv * scale[i]);
}

// ---------------------------------------------------------------- generic bf16 GEMM
// C[M,N] = A[M,K](bf16) x W[N,K](bf16)^T (+ add[M,N]) -> f32
// block = 256 = 8 waves (2x4) -> 32x64 tile.
// K loop unrolled x2 as a register ping-pong (K % 64 == 0 for all calls):
// fragment loads for stage s+1 are in flight while the WMMA for stage s runs.
__global__ void __launch_bounds__(256) gemm_bf16_nt_kernel(const unsigned short* __restrict__ A,
                                                           const unsigned short* __restrict__ W,
                                                           const float* __restrict__ addsrc,
                                                           float* __restrict__ C,
                                                           int M, int N, int K) {
  const int lane = threadIdx.x & 31;
  const int w    = threadIdx.x >> 5;
  const int hi   = lane >> 4;
  const int ml   = lane & 15;
  const int row0 = blockIdx.y * 32 + (w >> 2) * 16;
  const int col0 = blockIdx.x * 64 + (w & 3) * 16;

  const unsigned short* Ab = A + (size_t)(row0 + ml) * K + 16 * hi;  // A-frag base (k0=0)
  const unsigned short* Wb = W + (size_t)(col0 + ml) * K + 16 * hi;  // B-frag base (k0=0)

  v8f c = {};
  v16bf a0, b0, a1{}, b1{};
  load_a16(a0, Ab);
  load_b16(b0, Wb);
  for (int k0 = 0; k0 < K; k0 += 64) {
    load_a16(a1, Ab + k0 + 32);                 // stage 1 loads (always in range)
    load_b16(b1, Wb + k0 + 32);
    __builtin_prefetch(Ab + k0 + 128, 0, 3);    // global_prefetch_b8 (speculative)
    __builtin_prefetch(Wb + k0 + 128, 0, 3);
    c = __builtin_amdgcn_wmma_f32_16x16x32_bf16(false, a0, false, b0, (short)0, c, false, false);
    if (k0 + 64 < K) {                          // stage 0 loads for next iteration
      load_a16(a0, Ab + k0 + 64);
      load_b16(b0, Wb + k0 + 64);
    }
    c = __builtin_amdgcn_wmma_f32_16x16x32_bf16(false, a1, false, b1, (short)0, c, false, false);
  }
#pragma unroll
  for (int j = 0; j < 8; ++j) {
    const int r  = row0 + j + 8 * hi;            // C layout: VGPR j -> M = j + 8*hi
    const size_t idx = (size_t)r * N + col0 + ml;
    float v = c[j];
    if (addsrc) v += addsrc[idx];
    C[idx] = v;
  }
}

// ---------------------------------------------------------------- RoPE + head split -> bf16 [b,h,t,dh]
__global__ void __launch_bounds__(256) rope_split_kernel(const float* __restrict__ qkv,
                                                         unsigned short* __restrict__ qo,
                                                         unsigned short* __restrict__ ko,
                                                         unsigned short* __restrict__ vo) {
  const int row = blockIdx.x;                 // b*T + t
  const int b = row / T_, t = row % T_;
  const float* qr = qkv + (size_t)row * (3 * D_);
  const float LOG1E4 = 9.210340371976184f;    // ln(10000)
#pragma unroll
  for (int it = 0; it < 2; ++it) {
    const int p = threadIdx.x + 256 * it;     // 0..511 : H_*D2_ rope pairs
    const int h = p >> 5, i = p & 31;
    const float freq = __expf(-((float)(2 * i) / (float)DH_) * LOG1E4);
    float cs, sn; __sincosf((float)t * freq, &cs, &sn);
    const size_t ob = ((size_t)(b * H_ + h) * T_ + t) * DH_;
    const float q1 = qr[h * DH_ + i], q2 = qr[h * DH_ + D2_ + i];
    qo[ob + i]        = f2bf(q1 * cs - q2 * sn);
    qo[ob + D2_ + i]  = f2bf(q2 * cs + q1 * sn);
    const float k1 = qr[D_ + h * DH_ + i], k2 = qr[D_ + h * DH_ + D2_ + i];
    ko[ob + i]        = f2bf(k1 * cs - k2 * sn);
    ko[ob + D2_ + i]  = f2bf(k2 * cs + k1 * sn);
  }
#pragma unroll
  for (int j = 0; j < 4; ++j) {
    const int e = threadIdx.x + 256 * j;      // 0..1023
    const int h = e >> 6, i = e & 63;
    vo[((size_t)(b * H_ + h) * T_ + t) * DH_ + i] = f2bf(qr[2 * D_ + e]);
  }
}

// ---------------------------------------------------------------- causal flash attention (bf16 WMMA)
// grid = (T/64, B*H), block = 128 (4 waves); wave owns 16 query rows.
// Double-buffered async global->LDS staging of K/V tiles (32 keys x 64 dh),
// next tile's DMA overlaps current tile's WMMAs (s_wait_asynccnt 4).
__global__ void __launch_bounds__(128) attn_kernel(const unsigned short* __restrict__ Q,
                                                   const unsigned short* __restrict__ Kc,
                                                   const unsigned short* __restrict__ V,
                                                   unsigned short* __restrict__ Y) {
  const int lane = threadIdx.x & 31;
  const int w    = threadIdx.x >> 5;
  const int hi   = lane >> 4;
  const int ml   = lane & 15;
  const int bh   = blockIdx.y;
  const int qbase = blockIdx.x * 64 + w * 16;

  const unsigned short* Qb = Q  + (size_t)bh * T_ * DH_;
  const unsigned short* Kb = Kc + (size_t)bh * T_ * DH_;
  const unsigned short* Vb = V  + (size_t)bh * T_ * DH_;

  __shared__ unsigned short Kt[2][32 * DH_];    // 2 x 4 KB
  __shared__ unsigned short Vt[2][32 * DH_];    // 2 x 4 KB
  __shared__ unsigned short Pbuf[4][16 * 32];   // per-wave P staging
  unsigned short* P = Pbuf[w];

  // stage helper: each thread issues 4 async b128 loads per tile
  auto stage = [&](int key0, int buf) {
    const unsigned short* gK = Kb + (size_t)key0 * DH_;
    const unsigned short* gV = Vb + (size_t)key0 * DH_;
#pragma unroll
    for (int j = 0; j < 2; ++j) {
      const int off = threadIdx.x * 8 + j * 1024;
      cp_g2l_16B(gK + off, Kt[buf] + off);
      cp_g2l_16B(gV + off, Vt[buf] + off);
    }
  };

  // preload Q fragments (two dh halves of 32)
  v16bf aq[2];
#pragma unroll
  for (int kh = 0; kh < 2; ++kh)
    load_a16(aq[kh], Qb + (size_t)(qbase + ml) * DH_ + kh * 32 + 16 * hi);

  v8f o[4]; o[0] = v8f{}; o[1] = v8f{}; o[2] = v8f{}; o[3] = v8f{};
  float mrow[8], lrow[8];
#pragma unroll
  for (int j = 0; j < 8; ++j) { mrow[j] = -1e30f; lrow[j] = 0.f; }

  const int nkb = (blockIdx.x * 64 + 95) >> 5;  // block-uniform #key-blocks (max wave)
  stage(0, 0);                                  // prologue: tile 0 in flight
  for (int kb = 0; kb < nkb; ++kb) {
    const int key0 = kb * 32;
    const int buf  = kb & 1;
    if (kb + 1 < nkb) {
      stage(key0 + 32, buf ^ 1);                // issue next tile's DMA first
      wait_async_n<4>();                        // wait only for tile kb (in-order)
    } else {
      wait_async_n<0>();
    }
    __syncthreads();                            // tile kb visible block-wide

    if (key0 <= qbase + 15) {                   // this wave's causal band
      v8f s[2];
#pragma unroll
      for (int sub = 0; sub < 2; ++sub) {
        v8f sv = {};
#pragma unroll
        for (int kh = 0; kh < 2; ++kh) {
          v16bf bk;
          load_b16(bk, Kt[buf] + (sub * 16 + ml) * DH_ + kh * 32 + 16 * hi);
          sv = __builtin_amdgcn_wmma_f32_16x16x32_bf16(false, aq[kh], false, bk,
                                                       (short)0, sv, false, false);
        }
#pragma unroll
        for (int j = 0; j < 8; ++j) {
          float vv = sv[j] * 0.125f;            // 1/sqrt(64)
          const int key  = key0 + sub * 16 + ml;
          const int qrow = qbase + j + 8 * hi;
          if (key > qrow) vv = -1e9f;           // causal mask
          sv[j] = vv;
        }
        s[sub] = sv;
      }
      // online softmax update + stage P into LDS (C-layout -> A-layout reshape)
#pragma unroll
      for (int j = 0; j < 8; ++j) {
        float mx = fmaxf(s[0][j], s[1][j]);
        for (int d = 1; d < 16; d <<= 1) mx = fmaxf(mx, __shfl_xor(mx, d));
        const float mnew  = fmaxf(mrow[j], mx);
        const float alpha = __expf(mrow[j] - mnew);
        mrow[j] = mnew;
        const float p0 = __expf(s[0][j] - mnew);
        const float p1 = __expf(s[1][j] - mnew);
        float ls = p0 + p1;
        for (int d = 1; d < 16; d <<= 1) ls += __shfl_xor(ls, d);
        lrow[j] = lrow[j] * alpha + ls;
        o[0][j] *= alpha; o[1][j] *= alpha; o[2][j] *= alpha; o[3][j] *= alpha;
        const int m = j + 8 * hi;
        P[m * 32 + ml]      = f2bf(p0);
        P[m * 32 + 16 + ml] = f2bf(p1);
      }
      asm volatile("s_wait_dscnt 0" ::: "memory");  // in-wave LDS RAW across lanes
      v16bf apf;
      load_a16(apf, P + ml * 32 + 16 * hi);
      // O += P (16x32) x V (32x64), four 16-col tiles, V from LDS
#pragma unroll
      for (int nt = 0; nt < 4; ++nt) {
        v16bf bv;
        unsigned short* bvp = (unsigned short*)&bv;
#pragma unroll
        for (int e = 0; e < 16; ++e)
          bvp[e] = Vt[buf][(16 * hi + e) * DH_ + nt * 16 + ml];
        o[nt] = __builtin_amdgcn_wmma_f32_16x16x32_bf16(false, apf, false, bv,
                                                        (short)0, o[nt], false, false);
      }
    }
    __syncthreads();                            // tile buf may be overwritten next iter
  }

  // normalize + scatter to [b, t, h*dh] for the proj GEMM
  const int b = bh >> 4, h = bh & 15;
#pragma unroll
  for (int j = 0; j < 8; ++j) {
    const float invl = 1.f / lrow[j];
    const int trow = qbase + j + 8 * hi;
    const size_t base = ((size_t)b * T_ + trow) * D_ + h * DH_;
#pragma unroll
    for (int nt = 0; nt < 4; ++nt)
      Y[base + nt * 16 + ml] = f2bf(o[nt][j] * invl);
  }
}

// ---------------------------------------------------------------- SSM scan (constant complex decay)
__global__ void __launch_bounds__(256) ssm_scan_kernel(const float* __restrict__ lambda_raw,
                                                       const float* __restrict__ omega,
                                                       const float* __restrict__ uproj,
                                                       const float* __restrict__ res,
                                                       float* __restrict__ ssm) {
  const int b = blockIdx.x, r = threadIdx.x;
  const float lam = -5.f / (1.f + __expf(-lambda_raw[r]));   // -5*sigmoid
  const float mag = __expf(lam);
  float cs, sn; __sincosf(omega[r], &cs, &sn);
  const float ar = mag * cs, ai = mag * sn;
  float hr = 0.f, hic = 0.f;
  const float* up = uproj + (size_t)b * T_ * RS_;
  const float* rs = res   + (size_t)b * T_ * R2_;
  float*       so = ssm   + (size_t)b * T_ * R2_;
  for (int t = 0; t < T_; ++t) {
    const float u   = up[(size_t)t * RS_ + r];
    const float nhr = ar * hr - ai * hic + u;
    const float nhi = ar * hic + ai * hr;
    hr = nhr; hic = nhi;
    so[(size_t)t * R2_ + r]       = hr  + rs[(size_t)t * R2_ + r];
    so[(size_t)t * R2_ + RS_ + r] = hic + rs[(size_t)t * R2_ + RS_ + r];
  }
}

// ---------------------------------------------------------------- LayerNorm(512) -> bf16
__global__ void __launch_bounds__(256) layernorm_bf16_kernel(const float* __restrict__ in,
                                                             const float* __restrict__ gamma,
                                                             const float* __restrict__ beta,
                                                             unsigned short* __restrict__ out) {
  __shared__ float red[256];
  const int row = blockIdx.x;
  const float* ir = in + (size_t)row * R2_;
  const float v0 = ir[threadIdx.x], v1 = ir[threadIdx.x + 256];
  red[threadIdx.x] = v0 + v1; __syncthreads();
  for (int off = 128; off > 0; off >>= 1) {
    if (threadIdx.x < off) red[threadIdx.x] += red[threadIdx.x + off];
    __syncthreads();
  }
  const float mu = red[0] / (float)R2_;
  __syncthreads();
  red[threadIdx.x] = (v0 - mu) * (v0 - mu) + (v1 - mu) * (v1 - mu); __syncthreads();
  for (int off = 128; off > 0; off >>= 1) {
    if (threadIdx.x < off) red[threadIdx.x] += red[threadIdx.x + off];
    __syncthreads();
  }
  const float inv = rsqrtf(red[0] / (float)R2_ + 1e-5f);
  unsigned short* orow = out + (size_t)row * R2_;
  orow[threadIdx.x]       = f2bf((v0 - mu) * inv * gamma[threadIdx.x] + beta[threadIdx.x]);
  orow[threadIdx.x + 256] = f2bf((v1 - mu) * inv * gamma[threadIdx.x + 256] + beta[threadIdx.x + 256]);
}

// ================================================================ host
extern "C" void kernel_launch(void* const* d_in, const int* in_sizes, int n_in,
                              void* d_out, int out_size, void* d_ws, size_t ws_size,
                              hipStream_t stream) {
  (void)in_sizes; (void)n_in; (void)out_size; (void)ws_size;
  const float* x          = (const float*)d_in[0];
  const float* Wqkv       = (const float*)d_in[1];
  const float* Wproj      = (const float*)d_in[2];
  const float* rms1_scale = (const float*)d_in[3];
  const float* rms2_scale = (const float*)d_in[4];
  const float* lambda_raw = (const float*)d_in[5];
  const float* omega      = (const float*)d_in[6];
  const float* W_sproj    = (const float*)d_in[7];
  const float* W_res      = (const float*)d_in[8];
  const float* ln_gamma   = (const float*)d_in[9];
  const float* ln_beta    = (const float*)d_in[10];
  const float* W_out      = (const float*)d_in[11];
  float* out = (float*)d_out;

  char* ws = (char*)d_ws;
  size_t cur = 0;
  auto take = [&](size_t bytes) { size_t o = cur; cur = (cur + bytes + 255) & ~(size_t)255; return o; };

  unsigned short* wqkv16   = (unsigned short*)(ws + take((size_t)3 * D_ * D_ * 2));
  unsigned short* wproj16  = (unsigned short*)(ws + take((size_t)D_ * D_ * 2));
  unsigned short* wsproj16 = (unsigned short*)(ws + take((size_t)RS_ * D_ * 2));
  unsigned short* wres16   = (unsigned short*)(ws + take((size_t)R2_ * D_ * 2));
  unsigned short* wout16   = (unsigned short*)(ws + take((size_t)D_ * R2_ * 2));
  unsigned short* xn16     = (unsigned short*)(ws + take((size_t)BT_ * D_ * 2));
  float*          qkv      = (float*)         (ws + take((size_t)BT_ * 3 * D_ * 4));
  unsigned short* q16      = (unsigned short*)(ws + take((size_t)BT_ * D_ * 2));
  unsigned short* k16      = (unsigned short*)(ws + take((size_t)BT_ * D_ * 2));
  unsigned short* v16      = (unsigned short*)(ws + take((size_t)BT_ * D_ * 2));
  unsigned short* y16      = (unsigned short*)(ws + take((size_t)BT_ * D_ * 2));
  float*          x2       = (float*)         (ws + take((size_t)BT_ * D_ * 4));
  unsigned short* u16      = (unsigned short*)(ws + take((size_t)BT_ * D_ * 2));
  float*          uproj    = (float*)         (ws + take((size_t)BT_ * RS_ * 4));
  float*          resb     = (float*)         (ws + take((size_t)BT_ * R2_ * 4));
  float*          ssm      = (float*)         (ws + take((size_t)BT_ * R2_ * 4));
  unsigned short* ln16     = (unsigned short*)(ws + take((size_t)BT_ * R2_ * 2));

  // 1) weights -> bf16
  cvt_bf16_kernel<<<2048, 256, 0, stream>>>(Wqkv,    wqkv16,   3 * D_ * D_);
  cvt_bf16_kernel<<<1024, 256, 0, stream>>>(Wproj,   wproj16,  D_ * D_);
  cvt_bf16_kernel<<<256,  256, 0, stream>>>(W_sproj, wsproj16, RS_ * D_);
  cvt_bf16_kernel<<<512,  256, 0, stream>>>(W_res,   wres16,   R2_ * D_);
  cvt_bf16_kernel<<<512,  256, 0, stream>>>(W_out,   wout16,   D_ * R2_);

  // 2) RMSNorm1 -> bf16
  rmsnorm_bf16_kernel<<<BT_, 256, 0, stream>>>(x, rms1_scale, xn16, D_);

  // 3) QKV GEMM: [4096,3072] = xn * Wqkv^T
  gemm_bf16_nt_kernel<<<dim3(3 * D_ / 64, BT_ / 32), 256, 0, stream>>>(
      xn16, wqkv16, nullptr, qkv, BT_, 3 * D_, D_);

  // 4) RoPE + head split -> bf16 [b,h,t,dh]
  rope_split_kernel<<<BT_, 256, 0, stream>>>(qkv, q16, k16, v16);

  // 5) causal attention -> y [b,t,D] bf16
  attn_kernel<<<dim3(T_ / 64, B_ * H_), 128, 0, stream>>>(q16, k16, v16, y16);

  // 6) proj GEMM + residual: x2 = x + y * Wproj^T
  gemm_bf16_nt_kernel<<<dim3(D_ / 64, BT_ / 32), 256, 0, stream>>>(
      y16, wproj16, x, x2, BT_, D_, D_);

  // 7) RMSNorm2 -> u bf16
  rmsnorm_bf16_kernel<<<BT_, 256, 0, stream>>>(x2, rms2_scale, u16, D_);

  // 8) u_proj = u * W_sproj^T   [4096,256]
  gemm_bf16_nt_kernel<<<dim3(RS_ / 64, BT_ / 32), 256, 0, stream>>>(
      u16, wsproj16, nullptr, uproj, BT_, RS_, D_);

  // 9) res = u * W_res^T        [4096,512]
  gemm_bf16_nt_kernel<<<dim3(R2_ / 64, BT_ / 32), 256, 0, stream>>>(
      u16, wres16, nullptr, resb, BT_, R2_, D_);

  // 10) complex decay scan + concat + res add -> ssm [4096,512]
  ssm_scan_kernel<<<B_, RS_, 0, stream>>>(lambda_raw, omega, uproj, resb, ssm);

  // 11) LayerNorm(512) -> bf16
  layernorm_bf16_kernel<<<BT_, 256, 0, stream>>>(ssm, ln_gamma, ln_beta, ln16);

  // 12) out = x2 + ssm_ln * W_out^T
  gemm_bf16_nt_kernel<<<dim3(D_ / 64, BT_ / 32), 256, 0, stream>>>(
      ln16, wout16, x2, out, BT_, D_, R2_);
}
